// MultiheadCorrelation_76811195121908
// MI455X (gfx1250) — matrix operand order repaired and checked by reading, not compiled, not run on hardware
//
#include <hip/hip_runtime.h>
#include <hip/hip_bf16.h>
#include <math.h>

// Problem constants (reference: T=8192, B=4, D=1024, H=16)
#define T_LEN  8192
#define BATCH  4
#define DMODEL 1024
#define NHEADS 16
#define HDIM   64
#define MROWS  (T_LEN * BATCH)   // 32768 flattened (t,b) rows
#define N_QKV  (3 * DMODEL)      // 3072

typedef __bf16 bf16;
typedef __attribute__((ext_vector_type(8)))  __bf16 v8bf;
typedef __attribute__((ext_vector_type(16))) __bf16 v16bf;
typedef __attribute__((ext_vector_type(8)))  float  v8f;

// ---------------------------------------------------------------------------
// WMMA fragment helpers for V_WMMA_F32_16X16X32_BF16 (wave32).
// A (16xK=32, row-major K-contiguous source): lane L holds row L&15,
//   K = {koff..koff+7} U {16+koff..16+koff+7}, koff = (L>=16)?8:0.
// B (K=32 x 16, source stored N-major with K contiguous per row, i.e. W[n][k]):
//   lane L holds col L&15, K = {koff..koff+7} U {16+koff..16+koff+7}, koff = (L>=16)?8:0.
// ---------------------------------------------------------------------------
static __device__ __forceinline__ v16bf load_frag_a(const bf16* base, int ld, int lane) {
  const int r    = lane & 15;
  const int koff = (lane & 16) ? 8 : 0;
  const bf16* p  = base + r * ld + koff;
  v8bf lo = *(const v8bf*)(p);
  v8bf hi = *(const v8bf*)(p + 16);
  return __builtin_shufflevector(lo, hi, 0,1,2,3,4,5,6,7,8,9,10,11,12,13,14,15);
}

static __device__ __forceinline__ v16bf load_frag_b(const bf16* base, int ld, int lane) {
  const int c  = lane & 15;
  const int kg = (lane & 16) ? 8 : 0;
  const bf16* p = base + c * ld + kg;
  v8bf lo = *(const v8bf*)(p);
  v8bf hi = *(const v8bf*)(p + 16);
  return __builtin_shufflevector(lo, hi, 0,1,2,3,4,5,6,7,8,9,10,11,12,13,14,15);
}

static __device__ __forceinline__ v8f wmma_bf16f32(v16bf a, v16bf b, v8f c) {
  // (neg_a, A, neg_b, B, c_mod, C, reuse_a, reuse_b)
  return __builtin_amdgcn_wmma_f32_16x16x32_bf16(false, a, false, b, (short)0, c,
                                                 false, false);
}

// ---------------------------------------------------------------------------
// fp32 -> bf16 conversion (grid-stride)
// ---------------------------------------------------------------------------
__global__ __launch_bounds__(256) void cvt_f32_bf16(const float* __restrict__ x,
                                                    bf16* __restrict__ y, int n) {
  int i = blockIdx.x * blockDim.x + threadIdx.x;
  int stride = gridDim.x * blockDim.x;
  for (; i < n; i += stride) y[i] = (bf16)x[i];
}

// ---------------------------------------------------------------------------
// LDS-staged WMMA GEMM: C[M,N] = A[M,K] * W[N,K]^T + bias
// Block tile 64(M) x 128(N), BK=32, 8 waves (2x4), wave tile 32x32.
// M%64==0, N%128==0, K%32==0 guaranteed by the problem sizes.
// ---------------------------------------------------------------------------
template <bool OUT_F32>
__global__ __launch_bounds__(256) void gemm_bf16_kernel(
    const bf16* __restrict__ A, int lda,
    const bf16* __restrict__ W, int ldw,
    const float* __restrict__ bias,
    float* __restrict__ Cf, bf16* __restrict__ Cb, int ldc, int K) {
  __shared__ alignas(16) bf16 As[64 * 40];    // +8 pad per row
  __shared__ alignas(16) bf16 Bs[128 * 40];

  const int tid  = threadIdx.x;
  const int lane = tid & 31;
  const int wave = tid >> 5;
  const int m0   = blockIdx.y * 64;
  const int n0   = blockIdx.x * 128;
  const int m0w  = (wave & 1) * 32;
  const int n0w  = (wave >> 1) * 32;

  const v8f vzero = {0.f, 0.f, 0.f, 0.f, 0.f, 0.f, 0.f, 0.f};
  v8f acc[2][2];
  acc[0][0] = vzero; acc[0][1] = vzero; acc[1][0] = vzero; acc[1][1] = vzero;

  const int ar = tid >> 2, ac = (tid & 3) * 8;   // A slab: 64x32
  const int br = tid >> 1, bc = (tid & 1) * 16;  // B slab: 128x32
  const bf16* Ag = A + (size_t)(m0 + ar) * lda + ac;
  const bf16* Wg = W + (size_t)(n0 + br) * ldw + bc;

  for (int k0 = 0; k0 < K; k0 += 32) {
    *(v8bf*)&As[ar * 40 + ac]          = *(const v8bf*)(Ag + k0);
    *(v8bf*)&Bs[br * 40 + bc]          = *(const v8bf*)(Wg + k0);
    *(v8bf*)&Bs[br * 40 + bc + 8]      = *(const v8bf*)(Wg + k0 + 8);
    __syncthreads();

    v16bf a0 = load_frag_a(&As[(m0w)      * 40], 40, lane);
    v16bf a1 = load_frag_a(&As[(m0w + 16) * 40], 40, lane);
    v16bf b0 = load_frag_b(&Bs[(n0w)      * 40], 40, lane);
    v16bf b1 = load_frag_b(&Bs[(n0w + 16) * 40], 40, lane);
    acc[0][0] = wmma_bf16f32(a0, b0, acc[0][0]);
    acc[0][1] = wmma_bf16f32(a0, b1, acc[0][1]);
    acc[1][0] = wmma_bf16f32(a1, b0, acc[1][0]);
    acc[1][1] = wmma_bf16f32(a1, b1, acc[1][1]);
    __syncthreads();
  }

  const int rbase = (lane & 16) ? 8 : 0;
  const int cl    = lane & 15;
#pragma unroll
  for (int fi = 0; fi < 2; ++fi)
#pragma unroll
    for (int fj = 0; fj < 2; ++fj) {
      const int col = n0 + n0w + fj * 16 + cl;
      const float bv = bias[col];
#pragma unroll
      for (int v = 0; v < 8; ++v) {
        const int row = m0 + m0w + fi * 16 + rbase + v;
        const float val = acc[fi][fj][v] + bv;
        if (OUT_F32) Cf[(size_t)row * ldc + col] = val;
        else         Cb[(size_t)row * ldc + col] = (bf16)val;
      }
    }
}

// ---------------------------------------------------------------------------
// Per-(b,h) mean and centered sum-of-squares over T for q and k channels.
// mean/varsum layout: [bh][128] (0..63 = q channels, 64..127 = k channels)
// ---------------------------------------------------------------------------
__global__ __launch_bounds__(256) void meanvar_kernel(const bf16* __restrict__ qkv,
                                                      float* __restrict__ meanp,
                                                      float* __restrict__ varp) {
  const int bh = blockIdx.x;
  const int b = bh >> 4, h = bh & 15;
  const int tid  = threadIdx.x;
  const int c    = tid & 127;
  const int half = tid >> 7;
  const int col  = ((c >> 6) ? DMODEL : 0) + h * HDIM + (c & 63);
  const size_t rowstride = (size_t)BATCH * N_QKV;
  const bf16* p = qkv + (size_t)(half * (T_LEN / 2)) * rowstride + (size_t)b * N_QKV + col;

  float s = 0.f, ss = 0.f;
  for (int t = 0; t < T_LEN / 2; ++t) {
    float x = (float)p[(size_t)t * rowstride];
    s += x; ss += x * x;
  }
  __shared__ float shs[256];
  __shared__ float shq[256];
  shs[tid] = s; shq[tid] = ss;
  __syncthreads();
  if (half == 0) {
    s  += shs[tid + 128];
    ss += shq[tid + 128];
    const float m = s * (1.0f / (float)T_LEN);
    meanp[bh * 128 + c] = m;
    varp[bh * 128 + c]  = ss - (float)T_LEN * m * m;  // sum((x-m)^2)
  }
}

// ---------------------------------------------------------------------------
// Correlation: cov[i][j] = sum_t (q[t,i]-mq_i)(k[t,j]-mk_j) via WMMA over T,
// then corr = clip(cov/sqrt(sx*sy),0,1), softmax over i (per column j).
// Output stored TRANSPOSED: corrT[bh][j*64+i] (bf16) so the attn kernel's
// B-fragments are contiguous loads.
// 8 waves: 2-way T-split x 4 quadrants of the 64x64 output.
// ---------------------------------------------------------------------------
__global__ __launch_bounds__(256) void corr_kernel(const bf16* __restrict__ qkv,
                                                   const float* __restrict__ meanp,
                                                   const float* __restrict__ varp,
                                                   bf16* __restrict__ corrT) {
  __shared__ alignas(16) bf16 XcT[64 * 72];  // centered q^T chunk [i][t]
  __shared__ alignas(16) bf16 YcT[64 * 72];  // centered k^T chunk [j][t]
  __shared__ float meansh[128];
  __shared__ float vsh[128];
  __shared__ float covsh[64 * 65];

  const int bh = blockIdx.x;
  const int b = bh >> 4, h = bh & 15;
  const int tid = threadIdx.x;

  if (tid < 128) {
    meansh[tid] = meanp[bh * 128 + tid];
    vsh[tid]    = varp[bh * 128 + tid];
  }
  for (int i = tid; i < 64 * 65; i += 256) covsh[i] = 0.0f;
  __syncthreads();

  const int lane = tid & 31;
  const int wave = tid >> 5;
  const int ks   = wave >> 2;         // T sub-slice within chunk
  const int i0   = (wave & 1) * 32;   // output quadrant
  const int j0   = ((wave >> 1) & 1) * 32;
  const int tloc = ks * 32;

  const v8f vzero = {0.f, 0.f, 0.f, 0.f, 0.f, 0.f, 0.f, 0.f};
  v8f acc[2][2];
  acc[0][0] = vzero; acc[0][1] = vzero; acc[1][0] = vzero; acc[1][1] = vzero;

  const int tr = tid >> 2;            // 0..63 : t within chunk
  const int cg = (tid & 3) * 16;      // channel group
  const size_t rowstride = (size_t)BATCH * N_QKV;

  for (int tb = 0; tb < T_LEN; tb += 64) {
    const bf16* gq = qkv + (size_t)(tb + tr) * rowstride + (size_t)b * N_QKV + h * HDIM + cg;
    v8bf q0 = *(const v8bf*)gq;
    v8bf q1 = *(const v8bf*)(gq + 8);
    v8bf k0 = *(const v8bf*)(gq + DMODEL);
    v8bf k1 = *(const v8bf*)(gq + DMODEL + 8);
#pragma unroll
    for (int e = 0; e < 8; ++e) {
      XcT[(cg + e)     * 72 + tr] = (bf16)((float)q0[e] - meansh[cg + e]);
      XcT[(cg + 8 + e) * 72 + tr] = (bf16)((float)q1[e] - meansh[cg + 8 + e]);
      YcT[(cg + e)     * 72 + tr] = (bf16)((float)k0[e] - meansh[64 + cg + e]);
      YcT[(cg + 8 + e) * 72 + tr] = (bf16)((float)k1[e] - meansh[64 + cg + 8 + e]);
    }
    __syncthreads();

    v16bf a0 = load_frag_a(&XcT[(i0)      * 72 + tloc], 72, lane);
    v16bf a1 = load_frag_a(&XcT[(i0 + 16) * 72 + tloc], 72, lane);
    v16bf b0 = load_frag_b(&YcT[(j0)      * 72 + tloc], 72, lane);
    v16bf b1 = load_frag_b(&YcT[(j0 + 16) * 72 + tloc], 72, lane);
    acc[0][0] = wmma_bf16f32(a0, b0, acc[0][0]);
    acc[0][1] = wmma_bf16f32(a0, b1, acc[0][1]);
    acc[1][0] = wmma_bf16f32(a1, b0, acc[1][0]);
    acc[1][1] = wmma_bf16f32(a1, b1, acc[1][1]);
    __syncthreads();
  }

  // reduce the two T-split partials (and write quadrants) into LDS
  const int rbase = (lane & 16) ? 8 : 0;
  const int cl    = lane & 15;
#pragma unroll
  for (int fi = 0; fi < 2; ++fi)
#pragma unroll
    for (int fj = 0; fj < 2; ++fj)
#pragma unroll
      for (int v = 0; v < 8; ++v)
        atomicAdd(&covsh[(i0 + fi * 16 + rbase + v) * 65 + (j0 + fj * 16 + cl)],
                  acc[fi][fj][v]);
  __syncthreads();

  // clip + softmax over i (axis=-2), one thread per column j
  if (tid < 64) {
    const int j = tid;
    const float vyj = fmaxf(vsh[64 + j], 1e-20f);
    float r[64];
    float mx = -1e30f;
    for (int i = 0; i < 64; ++i) {
      const float c = covsh[i * 65 + j];
      const float d = sqrtf(fmaxf(vsh[i], 1e-20f) * vyj);
      float x = c / d;
      x = fminf(fmaxf(x, 0.0f), 1.0f);
      r[i] = x;
      mx = fmaxf(mx, x);
    }
    float s = 0.0f;
    for (int i = 0; i < 64; ++i) { const float e = expf(r[i] - mx); r[i] = e; s += e; }
    const float inv = 1.0f / s;
    for (int i = 0; i < 64; ++i)
      corrT[(size_t)bh * 4096 + j * 64 + i] = (bf16)(r[i] * inv);
  }
}

// ---------------------------------------------------------------------------
// attn[t,b,h*64+j] = sum_i v[t,b,h*64+i] * corr[i][j] ; corr stored as corrT.
// Grid: (T/128, B, H); wave w handles 16 consecutive t rows, full 64-wide j.
// ---------------------------------------------------------------------------
__global__ __launch_bounds__(256) void attn_kernel(const bf16* __restrict__ qkv,
                                                   const bf16* __restrict__ corrT,
                                                   bf16* __restrict__ attn) {
  const int t0   = blockIdx.x * 128;
  const int b    = blockIdx.y;
  const int h    = blockIdx.z;
  const int tid  = threadIdx.x;
  const int lane = tid & 31;
  const int wave = tid >> 5;
  const bf16* cb = corrT + (size_t)(b * NHEADS + h) * 4096;

  v16bf bfrag[2][4];
#pragma unroll
  for (int ksi = 0; ksi < 2; ++ksi)
#pragma unroll
    for (int nj = 0; nj < 4; ++nj)
      bfrag[ksi][nj] = load_frag_b(cb + (nj * 16) * 64 + ksi * 32, 64, lane);

  const size_t rowstride = (size_t)BATCH * N_QKV;
  const bf16* vbase = qkv + (size_t)(t0 + wave * 16) * rowstride +
                      (size_t)b * N_QKV + 2 * DMODEL + h * HDIM;

  const v8f vzero = {0.f, 0.f, 0.f, 0.f, 0.f, 0.f, 0.f, 0.f};
  v8f acc[4] = {vzero, vzero, vzero, vzero};
#pragma unroll
  for (int ksi = 0; ksi < 2; ++ksi) {
    v16bf a = load_frag_a(vbase + ksi * 32, (int)rowstride, lane);
#pragma unroll
    for (int nj = 0; nj < 4; ++nj) acc[nj] = wmma_bf16f32(a, bfrag[ksi][nj], acc[nj]);
  }

  const int rbase = (lane & 16) ? 8 : 0;
  const int cl    = lane & 15;
#pragma unroll
  for (int nj = 0; nj < 4; ++nj)
#pragma unroll
    for (int v = 0; v < 8; ++v) {
      const int t = t0 + wave * 16 + rbase + v;
      attn[(size_t)(t * BATCH + b) * DMODEL + h * HDIM + nj * 16 + cl] = (bf16)acc[nj][v];
    }
}

// ---------------------------------------------------------------------------
// Workspace layout (bytes), total ~345 MB
// ---------------------------------------------------------------------------
static constexpr size_t SZ_QBF  = (size_t)MROWS * DMODEL * 2;
static constexpr size_t SZ_WQKV = (size_t)N_QKV * DMODEL * 2;
static constexpr size_t SZ_WOUT = (size_t)DMODEL * DMODEL * 2;
static constexpr size_t SZ_QKV  = (size_t)MROWS * N_QKV * 2;
static constexpr size_t SZ_MEAN = (size_t)64 * 128 * 4;
static constexpr size_t SZ_VAR  = (size_t)64 * 128 * 4;
static constexpr size_t SZ_CORR = (size_t)64 * 4096 * 2;

static constexpr size_t OFF_QBF  = 0;
static constexpr size_t OFF_WQKV = OFF_QBF + SZ_QBF;
static constexpr size_t OFF_WOUT = OFF_WQKV + SZ_WQKV;
static constexpr size_t OFF_QKV  = OFF_WOUT + SZ_WOUT;
static constexpr size_t OFF_MEAN = OFF_QKV + SZ_QKV;
static constexpr size_t OFF_VAR  = OFF_MEAN + SZ_MEAN;
static constexpr size_t OFF_CORR = OFF_VAR + SZ_VAR;
static constexpr size_t OFF_ATTN = OFF_CORR + SZ_CORR;

extern "C" void kernel_launch(void* const* d_in, const int* in_sizes, int n_in,
                              void* d_out, int out_size, void* d_ws, size_t ws_size,
                              hipStream_t stream) {
  const float* query = (const float*)d_in[0];
  const float* Wqkv  = (const float*)d_in[1];
  const float* bqkv  = (const float*)d_in[2];
  const float* Wout  = (const float*)d_in[3];
  const float* bout  = (const float*)d_in[4];
  float* out = (float*)d_out;
  char* ws = (char*)d_ws;

  bf16* qbf    = (bf16*)(ws + OFF_QBF);
  bf16* wqkvbf = (bf16*)(ws + OFF_WQKV);
  bf16* woutbf = (bf16*)(ws + OFF_WOUT);
  bf16* qkvbf  = (bf16*)(ws + OFF_QKV);
  float* meanp = (float*)(ws + OFF_MEAN);
  float* varp  = (float*)(ws + OFF_VAR);
  bf16* corrT  = (bf16*)(ws + OFF_CORR);
  bf16* attnbf = (bf16*)(ws + OFF_ATTN);

  // 1) fp32 -> bf16 conversions
  {
    int n = MROWS * DMODEL;
    cvt_f32_bf16<<<dim3(8192), dim3(256), 0, stream>>>(query, qbf, n);
    n = N_QKV * DMODEL;
    cvt_f32_bf16<<<dim3((n + 1023) / 1024), dim3(256), 0, stream>>>(Wqkv, wqkvbf, n);
    n = DMODEL * DMODEL;
    cvt_f32_bf16<<<dim3((n + 1023) / 1024), dim3(256), 0, stream>>>(Wout, woutbf, n);
  }

  // 2) qkv = query @ Wqkv^T + bqkv   (M=32768, N=3072, K=1024) -> bf16
  gemm_bf16_kernel<false><<<dim3(N_QKV / 128, MROWS / 64), dim3(256), 0, stream>>>(
      qbf, DMODEL, wqkvbf, DMODEL, bqkv, nullptr, qkvbf, N_QKV, DMODEL);

  // 3) per-(b,h) channel means / centered sum-of-squares over T
  meanvar_kernel<<<dim3(BATCH * NHEADS), dim3(256), 0, stream>>>(qkvbf, meanp, varp);

  // 4) 64x64 covariance via WMMA over T + clip + column softmax -> corrT (bf16)
  corr_kernel<<<dim3(BATCH * NHEADS), dim3(256), 0, stream>>>(qkvbf, meanp, varp, corrT);

  // 5) attn = v @ corr  (per (b,h), K=64)
  attn_kernel<<<dim3(T_LEN / 128, BATCH, NHEADS), dim3(256), 0, stream>>>(qkvbf, corrT,
                                                                          attnbf);

  // 6) out = attn @ Wout^T + bout   (M=32768, N=1024, K=1024) -> f32
  gemm_bf16_kernel<true><<<dim3(DMODEL / 128, MROWS / 64), dim3(256), 0, stream>>>(
      attnbf, DMODEL, woutbf, DMODEL, bout, out, nullptr, DMODEL, DMODEL);
}
